// Transformer_86045374808929
// MI455X (gfx1250) — matrix operand order repaired and checked by reading, not compiled, not run on hardware
//
#include <hip/hip_runtime.h>

// ---------------------------------------------------------------------------
// Transformer forward (enc-dec, B=16, L=512, D=512, H=8, DK=64, DFF=2048,
// VOC=3000, 6+6 layers) for gfx1250 / MI455X.
// - All matmuls: v_wmma_f32_16x16x32_bf16 (f32 accumulate).
// - Weights converted + pre-transposed to bf16 [N][K] once per call, so GEMM
//   B-staging is a contiguous global->LDS tile copy -> done by the Tensor
//   Data Mover (tensor_load_to_lds + s_wait_tensorcnt) when available.
// - Fused flash attention (no LxL score materialization).
// - All WMMA fragments built from wide LDS loads (ds_load_b128 class).
// - Staging loads batched (load-all-then-convert) so they issue as clauses.
// ---------------------------------------------------------------------------

#define TOK     8192   // B*L
#define BSZ     16
#define LSEQ    512
#define NHEAD   8
#define DMODEL  512
#define DFFN    2048
#define NVOC    3000
#define NLAY    6

typedef __attribute__((ext_vector_type(16))) __bf16 v16bf;
typedef __attribute__((ext_vector_type(8)))  __bf16 v8bf;
typedef __attribute__((ext_vector_type(4)))  __bf16 v4bf;
typedef __attribute__((ext_vector_type(8)))  float  v8f;
typedef __attribute__((ext_vector_type(4)))  unsigned int u32x4;
typedef __attribute__((ext_vector_type(8)))  int          i32x8;
typedef __attribute__((ext_vector_type(4)))  int          i32x4;

#if defined(__gfx1250__) && __has_builtin(__builtin_amdgcn_tensor_load_to_lds)
#define USE_TDM 1
#else
#define USE_TDM 0
#endif

static __device__ __forceinline__ v8f wmma_bf16(v16bf a, v16bf b, v8f c) {
  // 8 args: (neg_a, A, neg_b, B, c_mod, C, reuse_a, reuse_b)
  return __builtin_amdgcn_wmma_f32_16x16x32_bf16(false, a, false, b, (short)0, c,
                                                 false, false);
}

static __device__ __forceinline__ v16bf cat16(v8bf lo, v8bf hi) {
  return __builtin_shufflevector(lo, hi, 0, 1, 2, 3, 4, 5, 6, 7,
                                 8, 9, 10, 11, 12, 13, 14, 15);
}

static __device__ __forceinline__ v4bf cvt4(float4 f) {
  v4bf r;
  r[0] = (__bf16)f.x; r[1] = (__bf16)f.y; r[2] = (__bf16)f.z; r[3] = (__bf16)f.w;
  return r;
}

// ---------------------------------------------------------------------------
// 1) weight convert + transpose:  out[n*K + k] = (bf16) in[k*N + n]
//    LDS-tiled 32x32 so both global reads and writes are coalesced.
// ---------------------------------------------------------------------------
__global__ __launch_bounds__(256) void transp_bf16_kernel(const float* __restrict__ in,
                                                          __bf16* __restrict__ out,
                                                          int K, int N) {
  __shared__ __bf16 tile[32][33];          // +1 pad: no bank conflicts
  const int tk0 = blockIdx.x * 32;
  const int tn0 = blockIdx.y * 32;
  const int r   = threadIdx.x >> 5;        // 0..7
  const int c   = threadIdx.x & 31;        // 0..31
#pragma unroll
  for (int j = 0; j < 4; ++j) {
    const int kk = r + 8 * j;
    float v = 0.f;
    if (tk0 + kk < K && tn0 + c < N) v = in[(size_t)(tk0 + kk) * N + tn0 + c];
    tile[kk][c] = (__bf16)v;
  }
  __syncthreads();
#pragma unroll
  for (int j = 0; j < 4; ++j) {
    const int nn = r + 8 * j;
    if (tn0 + nn < N && tk0 + c < K)
      out[(size_t)(tn0 + nn) * K + tk0 + c] = tile[c][nn];
  }
}

// ---------------------------------------------------------------------------
// 2) embedding gather + sinusoidal positional encoding (computed on the fly)
// ---------------------------------------------------------------------------
__global__ __launch_bounds__(256) void embed_kernel(const int* __restrict__ ids,
                                                    const float* __restrict__ emb,
                                                    float* __restrict__ x) {
  const int t   = blockIdx.x;           // token index 0..TOK-1
  const int pos = t & (LSEQ - 1);
  const int id  = ids[t];
  for (int c = threadIdx.x; c < DMODEL; c += 256) {
    const int   i    = c >> 1;
    const float freq = __expf(-(float)(2 * i) * (9.210340371976184f / (float)DMODEL));
    const float a    = (float)pos * freq;
    const float pe   = (c & 1) ? __cosf(a) : __sinf(a);
    x[(size_t)t * DMODEL + c] = emb[(size_t)id * DMODEL + c] + pe;
  }
}

// ---------------------------------------------------------------------------
// 3) WMMA GEMM:  C[M,N] = A[M,K](f32->bf16) * BT[N,K](bf16, pre-transposed),
//    optional ReLU. Block 256 thr = 8 waves; block tile 256x64; K chunks of 32.
//    Each wave owns 32 rows (2 A-frags) x 4 n-tiles -> 8 WMMAs per chunk,
//    every B-fragment reused for 2 WMMAs. B tile staged global->LDS by the
//    Tensor Data Mover when available (contiguous [n][k] copy, OOB rows
//    zero-filled by the descriptor's tensor_dim1 = N - n0).
//    Requires M%256==0, K%32==0, N%8==0.
// ---------------------------------------------------------------------------
__global__ __launch_bounds__(256) void gemm_kernel(const float* __restrict__ A,
                                                   const __bf16* __restrict__ BT,
                                                   float* __restrict__ C,
                                                   int M, int N, int K, int relu) {
  __shared__ __bf16 sA[256 * 32];    // 16 KB, row-major [m][k]
  __shared__ __bf16 sBT[64 * 32];    //  4 KB, [n][k]

  const int tid   = threadIdx.x;
  const int w     = tid >> 5;
  const int lane  = tid & 31;
  const int lhalf = lane >> 4;
  const int lmod  = lane & 15;
  const int m0    = blockIdx.x * 256;
  const int n0    = blockIdx.y * 64;

  v8f a0t0 = {}, a0t1 = {}, a0t2 = {}, a0t3 = {};   // rows w*32 + [0,16)
  v8f a1t0 = {}, a1t1 = {}, a1t2 = {}, a1t3 = {};   // rows w*32 + [16,32)

  for (int k0 = 0; k0 < K; k0 += 32) {
    // ---- stage A: batch all 8 float4 loads first (clause), then convert ----
    float4 fa[8];
#pragma unroll
    for (int j = 0; j < 8; ++j) {
      const int e = tid * 4 + j * 1024;
      fa[j] = *(const float4*)&A[(size_t)(m0 + (e >> 5)) * K + k0 + (e & 31)];
    }
#pragma unroll
    for (int j = 0; j < 8; ++j)
      *(v4bf*)&sA[tid * 4 + j * 1024] = cvt4(fa[j]);

    // ---- stage B tile (64 x 32 bf16, row stride K) ------------------------
#if USE_TDM
    if (tid < 32) {
      const unsigned ldsAddr = (unsigned)(size_t)&sBT[0];
      const unsigned long long ga =
          (unsigned long long)(size_t)BT +
          ((unsigned long long)n0 * (unsigned long long)K +
           (unsigned long long)k0) * 2ull;
      const unsigned td1 = (unsigned)(N - n0);      // rows beyond -> zero-fill
      u32x4 g0;
      g0[0] = 1u;                                    // count=1, user D#
      g0[1] = ldsAddr;                               // LDS byte address
      g0[2] = (unsigned)(ga & 0xFFFFFFFFull);        // global_addr[31:0]
      g0[3] = (unsigned)((ga >> 32) & 0x01FFFFFFull) // global_addr[56:32]
              | (2u << 30);                          // type = 2 (image)
      i32x8 g1;
      g1[0] = (int)(1u << 16);                       // data_size = 2 bytes
      g1[1] = (int)(32u << 16);                      // tensor_dim0 = 32
      g1[2] = (int)((td1 & 0xFFFFu) << 16);          // tensor_dim1 lo
      g1[3] = (int)((td1 >> 16) | (32u << 16));      // tensor_dim1 hi | tile_dim0=32
      g1[4] = 64;                                    // tile_dim1 = 64
      g1[5] = (int)(unsigned)K;                      // tensor_dim0_stride = K
      g1[6] = 0;
      g1[7] = 0;
      const i32x4 gz = {};
#if __clang_major__ >= 23
      const i32x8 gz8 = {};
      __builtin_amdgcn_tensor_load_to_lds(g0, g1, gz, gz, gz8, 0);
#else
      __builtin_amdgcn_tensor_load_to_lds(g0, g1, gz, gz, 0);
#endif
#if __has_builtin(__builtin_amdgcn_s_wait_tensorcnt)
      __builtin_amdgcn_s_wait_tensorcnt(0);
#endif
    }
#else
    {
      const int bn = tid >> 2;            // 0..63
      const int bk = (tid & 3) * 8;       // 0..24
      v8bf g = {};
      if (n0 + bn < N) g = *(const v8bf*)&BT[(size_t)(n0 + bn) * K + k0 + bk];
      *(v8bf*)&sBT[bn * 32 + bk] = g;
    }
#endif
    __syncthreads();

    if (k0 + 32 < K)
      __builtin_prefetch(&A[(size_t)(m0 + tid) * K + k0 + 32], 0, 3);

    // ---- A fragments: two contiguous 16B LDS reads each -------------------
    const int kb = lhalf * 16;
    const int ma = w * 32 + lmod;
    const int mb = ma + 16;
    const v16bf af0 = cat16(*(const v8bf*)&sA[ma * 32 + kb],
                            *(const v8bf*)&sA[ma * 32 + kb + 8]);
    const v16bf af1 = cat16(*(const v8bf*)&sA[mb * 32 + kb],
                            *(const v8bf*)&sA[mb * 32 + kb + 8]);

    // ---- B fragments: one contiguous 32B LDS read, reused for 2 WMMAs -----
    const int kb2 = lhalf * 16;
    {
      const v16bf bfr = *(const v16bf*)&sBT[(lmod) * 32 + kb2];
      a0t0 = wmma_bf16(af0, bfr, a0t0);
      a1t0 = wmma_bf16(af1, bfr, a1t0);
    }
    {
      const v16bf bfr = *(const v16bf*)&sBT[(16 + lmod) * 32 + kb2];
      a0t1 = wmma_bf16(af0, bfr, a0t1);
      a1t1 = wmma_bf16(af1, bfr, a1t1);
    }
    {
      const v16bf bfr = *(const v16bf*)&sBT[(32 + lmod) * 32 + kb2];
      a0t2 = wmma_bf16(af0, bfr, a0t2);
      a1t2 = wmma_bf16(af1, bfr, a1t2);
    }
    {
      const v16bf bfr = *(const v16bf*)&sBT[(48 + lmod) * 32 + kb2];
      a0t3 = wmma_bf16(af0, bfr, a0t3);
      a1t3 = wmma_bf16(af1, bfr, a1t3);
    }
    __syncthreads();
  }

  // ---- epilogue (C/D layout: m = i + 8*lhalf, n = lmod); coalesced stores --
  const int mrow = m0 + w * 32;
#pragma unroll
  for (int i = 0; i < 8; ++i) {
    const int ra = mrow + i + 8 * lhalf;
    const int rb = ra + 16;
    float v0 = a0t0[i], v1 = a0t1[i], v2 = a0t2[i], v3 = a0t3[i];
    float u0 = a1t0[i], u1 = a1t1[i], u2 = a1t2[i], u3 = a1t3[i];
    if (relu) {
      v0 = fmaxf(v0, 0.f); v1 = fmaxf(v1, 0.f); v2 = fmaxf(v2, 0.f); v3 = fmaxf(v3, 0.f);
      u0 = fmaxf(u0, 0.f); u1 = fmaxf(u1, 0.f); u2 = fmaxf(u2, 0.f); u3 = fmaxf(u3, 0.f);
    }
    float* ca = C + (size_t)ra * N + n0;
    float* cb = C + (size_t)rb * N + n0;
    if (n0 + lmod      < N) { ca[lmod]      = v0; cb[lmod]      = u0; }
    if (n0 + 16 + lmod < N) { ca[16 + lmod] = v1; cb[16 + lmod] = u1; }
    if (n0 + 32 + lmod < N) { ca[32 + lmod] = v2; cb[32 + lmod] = u2; }
    if (n0 + 48 + lmod < N) { ca[48 + lmod] = v3; cb[48 + lmod] = u3; }
  }
}

// ---------------------------------------------------------------------------
// 4) fused flash attention (one (b,h) x 64 q-rows per block; 4 waves x 16 q)
//    mask = key_ids[b,k]==0 (padding, key-only) | causal (k > q)
//    scores *= 1/sqrt(64), masked -> -1e9 (matches reference jnp.where)
// ---------------------------------------------------------------------------
__global__ __launch_bounds__(128) void attn_kernel(const float* __restrict__ Qm,
                                                   const float* __restrict__ Km,
                                                   const float* __restrict__ Vm,
                                                   float* __restrict__ Om,
                                                   const int* __restrict__ key_ids,
                                                   int causal) {
  __shared__ __bf16 sK[32 * 64];         // row-major [key][dk], 4 KB
  __shared__ __bf16 sVT[64 * 32];        // transposed [dk][key], 4 KB
  __shared__ __bf16 sP[4][16 * 32];      // wave-private P transpose, 4 KB

  const int tid   = threadIdx.x;
  const int w     = tid >> 5;
  const int lane  = tid & 31;
  const int lhalf = lane >> 4;
  const int lmod  = lane & 15;
  const int bh    = blockIdx.y;
  const int b     = bh >> 3;             // NHEAD == 8
  const int h     = bh & 7;
  const int qbase = blockIdx.x * 64 + w * 16;
  const int kb    = lhalf * 16;
  const int kb2   = lhalf * 16;

  // Q fragments (K-dim 0..31 / 32..63) from coalesced float4 loads
  v16bf aq0, aq1;
  {
    const int    qrow = qbase + lmod;
    const float* qp   = Qm + (size_t)(b * LSEQ + qrow) * DMODEL + h * 64;
    const v4bf a0 = cvt4(*(const float4*)(qp + kb));
    const v4bf a1 = cvt4(*(const float4*)(qp + kb + 4));
    const v4bf a2 = cvt4(*(const float4*)(qp + kb + 8));
    const v4bf a3 = cvt4(*(const float4*)(qp + kb + 12));
    aq0 = cat16(__builtin_shufflevector(a0, a1, 0, 1, 2, 3, 4, 5, 6, 7),
                __builtin_shufflevector(a2, a3, 0, 1, 2, 3, 4, 5, 6, 7));
    const v4bf b0 = cvt4(*(const float4*)(qp + 32 + kb));
    const v4bf b1 = cvt4(*(const float4*)(qp + 32 + kb + 4));
    const v4bf b2 = cvt4(*(const float4*)(qp + 32 + kb + 8));
    const v4bf b3 = cvt4(*(const float4*)(qp + 32 + kb + 12));
    aq1 = cat16(__builtin_shufflevector(b0, b1, 0, 1, 2, 3, 4, 5, 6, 7),
                __builtin_shufflevector(b2, b3, 0, 1, 2, 3, 4, 5, 6, 7));
  }

  v8f o0 = {}, o1 = {}, o2 = {}, o3 = {};
  float rmax[8], rsum[8];
#pragma unroll
  for (int i = 0; i < 8; ++i) { rmax[i] = -3.0e38f; rsum[i] = 0.f; }

  for (int kc = 0; kc < LSEQ; kc += 32) {
    // stage 32 keys/values: batch all float4 loads (clause), then convert;
    // K row-major (v4bf store), V transpose-scattered into [dk][key]
    float4 fk[4], fv[4];
#pragma unroll
    for (int j = 0; j < 4; ++j) {
      const int e = tid * 4 + j * 512;
      const int r = e >> 6, c = e & 63;
      const size_t g = (size_t)(b * LSEQ + kc + r) * DMODEL + h * 64 + c;
      fk[j] = *(const float4*)&Km[g];
      fv[j] = *(const float4*)&Vm[g];
    }
#pragma unroll
    for (int j = 0; j < 4; ++j) {
      const int e = tid * 4 + j * 512;
      const int r = e >> 6, c = e & 63;
      *(v4bf*)&sK[e] = cvt4(fk[j]);
      const v4bf vv = cvt4(fv[j]);
#pragma unroll
      for (int t = 0; t < 4; ++t) sVT[(c + t) * 32 + r] = vv[t];
    }
    __syncthreads();

    // S = Q*K^T : B-fragments are contiguous 32B reads from sK rows
    v8f s0 = {}, s1 = {};
    {
      const __bf16* kp0 = sK + lmod * 64;          // keys kc+0..15
      const __bf16* kp1 = sK + (16 + lmod) * 64;   // keys kc+16..31
      s0 = wmma_bf16(aq0, *(const v16bf*)(kp0 + kb2), s0);
      s0 = wmma_bf16(aq1, *(const v16bf*)(kp0 + 32 + kb2), s0);
      s1 = wmma_bf16(aq0, *(const v16bf*)(kp1 + kb2), s1);
      s1 = wmma_bf16(aq1, *(const v16bf*)(kp1 + 32 + kb2), s1);
    }

    // scale, mask, online softmax (xor reductions stay inside 16-lane halves)
    float p0[8], p1[8], corr[8];
    const int k0g  = kc + lmod;
    const int k1g  = kc + 16 + lmod;
    const int pad0 = (key_ids[b * LSEQ + k0g] == 0);
    const int pad1 = (key_ids[b * LSEQ + k1g] == 0);
#pragma unroll
    for (int i = 0; i < 8; ++i) {
      const int qm = qbase + i + 8 * lhalf;
      float v0 = s0[i] * 0.125f;
      float v1 = s1[i] * 0.125f;
      if (pad0 || (causal && k0g > qm)) v0 = -1e9f;
      if (pad1 || (causal && k1g > qm)) v1 = -1e9f;
      float cm = fmaxf(v0, v1);
#pragma unroll
      for (int off = 1; off < 16; off <<= 1) cm = fmaxf(cm, __shfl_xor(cm, off, 32));
      const float nm = fmaxf(rmax[i], cm);
      corr[i] = __expf(rmax[i] - nm);
      p0[i]   = __expf(v0 - nm);
      p1[i]   = __expf(v1 - nm);
      float cs = p0[i] + p1[i];
#pragma unroll
      for (int off = 1; off < 16; off <<= 1) cs += __shfl_xor(cs, off, 32);
      rsum[i] = rsum[i] * corr[i] + cs;
      rmax[i] = nm;
    }
#pragma unroll
    for (int i = 0; i < 8; ++i) {
      o0[i] *= corr[i]; o1[i] *= corr[i]; o2[i] *= corr[i]; o3[i] *= corr[i];
    }

    // transpose P (C-layout -> A-layout) through wave-private LDS
    __bf16* pw = sP[w];
#pragma unroll
    for (int i = 0; i < 8; ++i) {
      const int m = i + 8 * lhalf;
      pw[m * 32 + lmod]      = (__bf16)p0[i];
      pw[m * 32 + 16 + lmod] = (__bf16)p1[i];
    }
    const v16bf ap = cat16(*(const v8bf*)&pw[lmod * 32 + kb],
                           *(const v8bf*)&pw[lmod * 32 + kb + 8]);

    // O += P * V : B-fragments are contiguous 32B reads from sVT
    o0 = wmma_bf16(ap, *(const v16bf*)&sVT[(lmod)      * 32 + kb2], o0);
    o1 = wmma_bf16(ap, *(const v16bf*)&sVT[(16 + lmod) * 32 + kb2], o1);
    o2 = wmma_bf16(ap, *(const v16bf*)&sVT[(32 + lmod) * 32 + kb2], o2);
    o3 = wmma_bf16(ap, *(const v16bf*)&sVT[(48 + lmod) * 32 + kb2], o3);
    __syncthreads();
  }

  // normalize and write ctx back in [T, H*DK] layout (coalesced across lanes)
#pragma unroll
  for (int i = 0; i < 8; ++i) {
    const int   qrow = qbase + i + 8 * lhalf;
    const float inv  = 1.f / rsum[i];
    float* op = Om + (size_t)(b * LSEQ + qrow) * DMODEL + h * 64;
    op[lmod]      = o0[i] * inv;
    op[16 + lmod] = o1[i] * inv;
    op[32 + lmod] = o2[i] * inv;
    op[48 + lmod] = o3[i] * inv;
  }
}

// ---------------------------------------------------------------------------
// 5) out = layernorm(a + res), row width 512; one wave32 per row
// ---------------------------------------------------------------------------
__global__ __launch_bounds__(256) void add_ln_kernel(const float* a,
                                                     const float* res,
                                                     float* out, int rows) {
  const int w    = threadIdx.x >> 5;
  const int lane = threadIdx.x & 31;
  const int row  = blockIdx.x * 8 + w;
  if (row >= rows) return;

  const float* pa = a   + (size_t)row * DMODEL;
  const float* pr = res + (size_t)row * DMODEL;
  float v[16];
  float s = 0.f;
#pragma unroll
  for (int i = 0; i < 16; ++i) {
    v[i] = pa[lane + i * 32] + pr[lane + i * 32];
    s += v[i];
  }
#pragma unroll
  for (int off = 16; off; off >>= 1) s += __shfl_xor(s, off, 32);
  const float mu = s * (1.f / (float)DMODEL);
  float q = 0.f;
#pragma unroll
  for (int i = 0; i < 16; ++i) { const float d = v[i] - mu; q += d * d; }
#pragma unroll
  for (int off = 16; off; off >>= 1) q += __shfl_xor(q, off, 32);
  const float inv = rsqrtf(q * (1.f / (float)DMODEL) + 1e-5f);
  float* po = out + (size_t)row * DMODEL;
#pragma unroll
  for (int i = 0; i < 16; ++i) po[lane + i * 32] = (v[i] - mu) * inv;
}

// ---------------------------------------------------------------------------
// host orchestration
// ---------------------------------------------------------------------------
extern "C" void kernel_launch(void* const* d_in, const int* in_sizes, int n_in,
                              void* d_out, int out_size, void* d_ws, size_t ws_size,
                              hipStream_t stream) {
  (void)in_sizes; (void)n_in; (void)out_size; (void)ws_size;

  const int*   enc_ids = (const int*)d_in[0];
  const int*   dec_ids = (const int*)d_in[1];
  const float* src_emb = (const float*)d_in[2];
  const float* tgt_emb = (const float*)d_in[3];
  const float* proj_W  = (const float*)d_in[4];
  // 0..3: enc Wq/Wk/Wv/Wo ; 4..7: dec self ; 8..11: dec cross (each [NL,D,D])
  const float* Wf[16];
  for (int i = 0; i < 16; ++i) Wf[i] = (const float*)d_in[5 + i];
  // Wf[12]=enc_W1 [NL,D,DFF], Wf[13]=enc_W2 [NL,DFF,D], Wf[14]=dec_W1, Wf[15]=dec_W2

  // ---- workspace carve-up -------------------------------------------------
  char* base = (char*)d_ws;
  size_t off = 0;
  auto take = [&](size_t bytes) -> char* {
    char* r = base + off;
    off = (off + bytes + 255) & ~(size_t)255;
    return r;
  };
  const size_t WDD = (size_t)DMODEL * DMODEL;
  const size_t WDF = (size_t)DMODEL * DFFN;
  const size_t NDD = (size_t)NLAY * WDD;
  const size_t NDF = (size_t)NLAY * WDF;

  __bf16* bW[12];                                // transposed [N][K] per layer
  for (int i = 0; i < 12; ++i) bW[i] = (__bf16*)take(NDD * 2);
  __bf16* bW1e  = (__bf16*)take(NDF * 2);
  __bf16* bW2e  = (__bf16*)take(NDF * 2);
  __bf16* bW1d  = (__bf16*)take(NDF * 2);
  __bf16* bW2d  = (__bf16*)take(NDF * 2);
  __bf16* bProj = (__bf16*)take((size_t)DMODEL * NVOC * 2);

  const size_t ACT = (size_t)TOK * DMODEL * 4;
  float* x   = (float*)take(ACT);   // encoder stream (becomes enc_out)
  float* y   = (float*)take(ACT);   // decoder stream
  float* q   = (float*)take(ACT);
  float* k   = (float*)take(ACT);
  float* v   = (float*)take(ACT);
  float* ctx = (float*)take(ACT);
  float* tmp = (float*)take(ACT);
  float* hid = (float*)take((size_t)TOK * DFFN * 4);

  // ---- helpers ------------------------------------------------------------
  auto transp = [&](const float* src, __bf16* dst, int K, int N) {
    dim3 g((K + 31) / 32, (N + 31) / 32);
    hipLaunchKernelGGL(transp_bf16_kernel, g, dim3(256), 0, stream, src, dst, K, N);
  };
  auto gemm = [&](const float* A, const __bf16* Bm, float* Cm,
                  int M, int N, int K, int relu) {
    dim3 g(M / 256, (N + 63) / 64);
    hipLaunchKernelGGL(gemm_kernel, g, dim3(256), 0, stream, A, Bm, Cm, M, N, K, relu);
  };
  auto attn = [&](const float* Qm, const float* Km, const float* Vm, float* Om,
                  const int* ids, int causal) {
    dim3 g(LSEQ / 64, BSZ * NHEAD);
    hipLaunchKernelGGL(attn_kernel, g, dim3(128), 0, stream, Qm, Km, Vm, Om, ids, causal);
  };
  auto add_ln = [&](const float* a, const float* r, float* o) {
    hipLaunchKernelGGL(add_ln_kernel, dim3(TOK / 8), dim3(256), 0, stream, a, r, o, TOK);
  };

  // ---- weight convert + transpose (every call; deterministic) -------------
  for (int i = 0; i < 12; ++i)
    for (int l = 0; l < NLAY; ++l)
      transp(Wf[i] + l * WDD, bW[i] + l * WDD, DMODEL, DMODEL);
  for (int l = 0; l < NLAY; ++l) {
    transp(Wf[12] + l * WDF, bW1e + l * WDF, DMODEL, DFFN);
    transp(Wf[13] + l * WDF, bW2e + l * WDF, DFFN, DMODEL);
    transp(Wf[14] + l * WDF, bW1d + l * WDF, DMODEL, DFFN);
    transp(Wf[15] + l * WDF, bW2d + l * WDF, DFFN, DMODEL);
  }
  transp(proj_W, bProj, DMODEL, NVOC);

  // ---- encoder ------------------------------------------------------------
  hipLaunchKernelGGL(embed_kernel, dim3(TOK), dim3(256), 0, stream, enc_ids, src_emb, x);
  for (int l = 0; l < NLAY; ++l) {
    gemm(x, bW[0] + l * WDD, q, TOK, DMODEL, DMODEL, 0);
    gemm(x, bW[1] + l * WDD, k, TOK, DMODEL, DMODEL, 0);
    gemm(x, bW[2] + l * WDD, v, TOK, DMODEL, DMODEL, 0);
    attn(q, k, v, ctx, enc_ids, /*causal=*/0);
    gemm(ctx, bW[3] + l * WDD, tmp, TOK, DMODEL, DMODEL, 0);
    add_ln(tmp, x, x);
    gemm(x, bW1e + l * WDF, hid, TOK, DFFN, DMODEL, /*relu=*/1);
    gemm(hid, bW2e + l * WDF, tmp, TOK, DMODEL, DFFN, 0);
    add_ln(tmp, x, x);
  }
  // x == enc_out from here on

  // ---- decoder ------------------------------------------------------------
  hipLaunchKernelGGL(embed_kernel, dim3(TOK), dim3(256), 0, stream, dec_ids, tgt_emb, y);
  for (int l = 0; l < NLAY; ++l) {
    // masked self-attention
    gemm(y, bW[4] + l * WDD, q, TOK, DMODEL, DMODEL, 0);
    gemm(y, bW[5] + l * WDD, k, TOK, DMODEL, DMODEL, 0);
    gemm(y, bW[6] + l * WDD, v, TOK, DMODEL, DMODEL, 0);
    attn(q, k, v, ctx, dec_ids, /*causal=*/1);
    gemm(ctx, bW[7] + l * WDD, tmp, TOK, DMODEL, DMODEL, 0);
    add_ln(tmp, y, y);
    // cross-attention (K,V from enc_out)
    gemm(y, bW[8] + l * WDD, q, TOK, DMODEL, DMODEL, 0);
    gemm(x, bW[9] + l * WDD, k, TOK, DMODEL, DMODEL, 0);
    gemm(x, bW[10] + l * WDD, v, TOK, DMODEL, DMODEL, 0);
    attn(q, k, v, ctx, enc_ids, /*causal=*/0);
    gemm(ctx, bW[11] + l * WDD, tmp, TOK, DMODEL, DMODEL, 0);
    add_ln(tmp, y, y);
    // FFN
    gemm(y, bW1d + l * WDF, hid, TOK, DFFN, DMODEL, /*relu=*/1);
    gemm(hid, bW2d + l * WDF, tmp, TOK, DMODEL, DFFN, 0);
    add_ln(tmp, y, y);
  }

  // ---- output projection: logits [TOK, VOC] f32 ---------------------------
  gemm(y, bProj, (float*)d_out, TOK, NVOC, DMODEL, 0);
}
